// DilatedAttention_4990751998413
// MI455X (gfx1250) — matrix operand order repaired and checked by reading, not compiled, not run on hardware
//
#include <hip/hip_runtime.h>
#include <hip/hip_bf16.h>

typedef __attribute__((ext_vector_type(16))) _Float16 v16h;
typedef __attribute__((ext_vector_type(8)))  _Float16 v8h;
typedef __attribute__((ext_vector_type(4)))  _Float16 v4h;
typedef __attribute__((ext_vector_type(8)))  float    v8f;
typedef __attribute__((ext_vector_type(4)))  unsigned int u32x4;
typedef __attribute__((ext_vector_type(8)))  int          i32x8;
typedef __attribute__((ext_vector_type(4)))  int          i32x4;

#define DM     2048
#define QKV_N  6144
#define NROWS  8192      // B * N_SEG * L  (dilated tokens)
#define HD     128
#define LSEG   256
#define QB     64
#define LD_AB  40        // f16 stride for 32-wide K tiles (64B row + 16B pad)
#define BUFH   (128 * LD_AB)   // halves per LDS tile buffer
#define LDQ    136       // f16 stride for 128-wide rows (256B row + 16B pad)
#define LDS_S  260       // f32 stride for 256-wide score rows
#define LDP    264       // f16 stride for 256-long K dimension

#define HAVE_TDM __has_builtin(__builtin_amdgcn_tensor_load_to_lds)

union V16U { v16h v; v8h h[2]; };

// A fragment (16x32 f16): M=laneLo; K halves {laneHi*8..+7, 16+laneHi*8..+7}
__device__ __forceinline__ v16h frag_a(const _Float16* rowptr_k0, int laneHi) {
    V16U u;
    u.h[0] = *(const v8h*)(rowptr_k0 + laneHi * 8);
    u.h[1] = *(const v8h*)(rowptr_k0 + 16 + laneHi * 8);
    return u.v;
}
// B fragment (32x16 f16): column N=laneLo; K halves {laneHi*8..+7, 16+laneHi*8..+7}
__device__ __forceinline__ v16h frag_b(const _Float16* colptr_k0, int laneHi) {
    V16U u;
    u.h[0] = *(const v8h*)(colptr_k0 + laneHi * 8);
    u.h[1] = *(const v8h*)(colptr_k0 + 16 + laneHi * 8);
    return u.v;
}
__device__ __forceinline__ v8f wmma16(v16h a, v16h b, v8f c) {
    return __builtin_amdgcn_wmma_f32_16x16x32_f16(false, a, false, b, (short)0, c,
                                                  false, false);
}

__device__ __forceinline__ unsigned lds_addr_u32(const void* p) {
    return (unsigned)(unsigned long long)(uintptr_t)p;
}

#if HAVE_TDM
// TDM: DMA a [tileRows x tileK] f16 tile (row stride ldRow elems) into LDS,
// inserting padAmount(+1) DWORDs of padding every 2<<padInterval DWORDs.
__device__ __forceinline__ void tdm_load_tile(unsigned ldsAddr, const _Float16* g,
                                              int tileK, int tileRows, int ldRow,
                                              int padInterval, int padAmount) {
    unsigned long long ga = (unsigned long long)(uintptr_t)g;
    u32x4 g0;
    g0[0] = 1u;                                           // count=1
    g0[1] = ldsAddr;                                      // lds_addr
    g0[2] = (unsigned)ga;                                 // global_addr[31:0]
    g0[3] = (unsigned)((ga >> 32) & 0x01ffffffu) | (2u << 30);  // addr[56:32]|type=2
    unsigned d0 = (unsigned)ldRow;                        // tensor_dim0
    unsigned d1 = 32768u;                                 // tensor_dim1 (>= tileRows)
    i32x8 g1;
    g1[0] = (int)((1u << 16) | (1u << 20) |               // data_size=2B, pad_enable
                  ((unsigned)padInterval << 22) | ((unsigned)padAmount << 25));
    g1[1] = (int)((d0 & 0xffffu) << 16);                  // tensor_dim0[15:0]
    g1[2] = (int)(((d0 >> 16) & 0xffffu) | ((d1 & 0xffffu) << 16));
    g1[3] = (int)(((d1 >> 16) & 0xffffu) | ((unsigned)tileK << 16)); // tile_dim0
    g1[4] = (int)((unsigned)tileRows & 0xffffu);          // tile_dim1, tile_dim2=0
    g1[5] = (int)(unsigned)ldRow;                         // tensor_dim0_stride[31:0]
    g1[6] = 0;
    g1[7] = 0;
    i32x4 z4 = {0, 0, 0, 0};
    i32x8 z8 = {0, 0, 0, 0, 0, 0, 0, 0};
    __builtin_amdgcn_tensor_load_to_lds(g0, g1, z4, z4, z8, 0);
}
#endif

// ---------------------------------------------------------------------------
// Elementwise fp32 -> f16 pre-passes (one-time conversion, removes all cvt
// VALU + half the global bytes from the hot GEMM loops)
// ---------------------------------------------------------------------------
__global__ __launch_bounds__(256)
void cvt_gather_x(const float* __restrict__ x, _Float16* __restrict__ xs16) {
    const long g = (long)blockIdx.x * 256 + threadIdx.x;  // one float4 each
    const int row = (int)(g >> 9);
    const int c = ((int)g & 511) * 4;
    const int bb = row >> 12, nn = (row >> 8) & 15, ll = row & 255;
    const float4 f =
        *(const float4*)(x + ((long)bb * 8192 + nn * 512 + ll * 2) * DM + c);
    *(v4h*)(xs16 + (long)row * DM + c) =
        (v4h){(_Float16)f.x, (_Float16)f.y, (_Float16)f.z, (_Float16)f.w};
}

__global__ __launch_bounds__(256)
void cvt_f32_f16(const float* __restrict__ src, _Float16* __restrict__ dst) {
    const long g = (long)blockIdx.x * 256 + threadIdx.x;
    const float4 f = *(const float4*)(src + g * 4);
    *(v4h*)(dst + g * 4) =
        (v4h){(_Float16)f.x, (_Float16)f.y, (_Float16)f.z, (_Float16)f.w};
}

// ---------------------------------------------------------------------------
// Shared 128x128 GEMM core: C += A(128xK) * B(128xK)^T, f16 in, f32 acc.
// TDM path: wave0 drives double-buffered tensor_load_to_lds, overlapped with
// WMMA via s_wait_tensorcnt. Buffer selected by INTEGER offset so fragment
// reads keep LDS provenance (ds_load_b128, not flat). Fallback: f16 copies.
// ---------------------------------------------------------------------------
__device__ __forceinline__ void do_wmma_block(const _Float16* cA, const _Float16* cB,
                                              int waveM, int waveN, int laneLo,
                                              int laneHi, v8f acc[2][4]) {
    v16h a[2], b[4];
#pragma unroll
    for (int im = 0; im < 2; ++im)
        a[im] = frag_a(cA + (waveM * 32 + im * 16 + laneLo) * LD_AB, laneHi);
#pragma unroll
    for (int jn = 0; jn < 4; ++jn)
        b[jn] = frag_b(cB + (waveN * 64 + jn * 16 + laneLo) * LD_AB, laneHi);
#pragma unroll
    for (int im = 0; im < 2; ++im)
#pragma unroll
        for (int jn = 0; jn < 4; ++jn)
            acc[im][jn] = wmma16(a[im], b[jn], acc[im][jn]);
}

__device__ __forceinline__ void gemm_tiles_f16(const _Float16* __restrict__ Abase,
                                               const _Float16* __restrict__ Bbase,
                                               int lda, int ldb, int kTotal,
                                               _Float16* sA, _Float16* sB,
                                               v8f acc[2][4]) {
    const int tid = threadIdx.x;
    const int lane = tid & 31, wave = tid >> 5;
    const int laneLo = lane & 15, laneHi = lane >> 4;
    const int waveM = wave >> 1, waveN = wave & 1;
#if HAVE_TDM
    const int nChunks = kTotal >> 5;
    const unsigned ldsA = lds_addr_u32(sA);
    const unsigned ldsB = lds_addr_u32(sB);
    if (wave == 0) {
        tdm_load_tile(ldsA, Abase, 32, 128, lda, 3, 3);
        tdm_load_tile(ldsB, Bbase, 32, 128, ldb, 3, 3);
    }
    for (int i = 0; i < nChunks; ++i) {
        const int curH = (i & 1) * BUFH;          // halves
        const int nxtH = BUFH - curH;
        if (wave == 0) {
            if (i + 1 < nChunks) {
                tdm_load_tile(ldsA + (unsigned)nxtH * 2, Abase + (i + 1) * 32,
                              32, 128, lda, 3, 3);
                tdm_load_tile(ldsB + (unsigned)nxtH * 2, Bbase + (i + 1) * 32,
                              32, 128, ldb, 3, 3);
                __builtin_amdgcn_s_wait_tensorcnt(2);   // oldest pair done
            } else {
                __builtin_amdgcn_s_wait_tensorcnt(0);
            }
        }
        __syncthreads();                                // publish buf[cur]
        do_wmma_block(sA + curH, sB + curH, waveM, waveN, laneLo, laneHi, acc);
        __syncthreads();                                // done reading buf[cur]
    }
#else
    const int arow = tid >> 1;
    const int off = (tid & 1) * 16;
    for (int k0 = 0; k0 < kTotal; k0 += 32) {
        const _Float16* ga = Abase + (long)arow * lda + k0 + off;
        const _Float16* gb = Bbase + (long)arow * ldb + k0 + off;
        *(v8h*)(sA + arow * LD_AB + off)     = *(const v8h*)(ga);
        *(v8h*)(sA + arow * LD_AB + off + 8) = *(const v8h*)(ga + 8);
        *(v8h*)(sB + arow * LD_AB + off)     = *(const v8h*)(gb);
        *(v8h*)(sB + arow * LD_AB + off + 8) = *(const v8h*)(gb + 8);
        __syncthreads();
        do_wmma_block(sA, sB, waveM, waveN, laneLo, laneHi, acc);
        __syncthreads();
    }
#endif
}

// ---------------------------------------------------------------------------
// Kernel 1: qkv = xs16 @ Wqkv^T + b, scattered to per-head f16 q/k/v ws
// ---------------------------------------------------------------------------
__global__ __launch_bounds__(256)
void qkv_gemm(const _Float16* __restrict__ xs16, const _Float16* __restrict__ w16,
              const float* __restrict__ bqkv, _Float16* __restrict__ q16,
              _Float16* __restrict__ k16, _Float16* __restrict__ v16) {
    __shared__ _Float16 sA[2 * BUFH];
    __shared__ _Float16 sB[2 * BUFH];
    const int tid = threadIdx.x;
    const int lane = tid & 31, wave = tid >> 5;
    const int laneLo = lane & 15, laneHi = lane >> 4;
    const int waveM = wave >> 1, waveN = wave & 1;
    const int mTile = blockIdx.y * 128, nTile = blockIdx.x * 128;

    v8f acc[2][4] = {};
    gemm_tiles_f16(xs16 + (long)mTile * DM, w16 + (long)nTile * DM, DM, DM, DM,
                   sA, sB, acc);

#pragma unroll
    for (int im = 0; im < 2; ++im) {
        const int m0 = mTile + waveM * 32 + im * 16 + laneHi * 8;
        const int bb = m0 >> 12, nn = (m0 >> 8) & 15, ll = m0 & 255;
#pragma unroll
        for (int jn = 0; jn < 4; ++jn) {
            const int e = nTile + waveN * 64 + jn * 16 + laneLo;
            const int which = e >> 11;
            const int d = e & 2047;
            const int h = d >> 7, c = d & 127;
            const float bias = bqkv[e];
            _Float16* dst = (which == 0) ? q16 : ((which == 1) ? k16 : v16);
            const long base = ((((long)bb * 16 + nn) * 16 + h) * 256 + ll) * 128 + c;
#pragma unroll
            for (int i = 0; i < 8; ++i)
                dst[base + (long)i * 128] = (_Float16)(acc[im][jn][i] + bias);
        }
    }
}

// ---------------------------------------------------------------------------
// Kernel 2: per-(b,seg,head,qblock) attention (f16 q/k/v, f16 out)
// ---------------------------------------------------------------------------
#define SM_Q  0
#define SM_K  17408                      // 64*136*2
#define SM_S  (SM_K + 69632)             // 256*136*2
#define SM_P  (SM_S + 66560)             // 64*260*4
#define SM_SZ (SM_P + 33792)             // + 64*264*2  = 187392 bytes

__global__ __launch_bounds__(256)
void attn_kernel(const _Float16* __restrict__ q16, const _Float16* __restrict__ k16,
                 const _Float16* __restrict__ v16, _Float16* __restrict__ o16) {
    extern __shared__ char smem[];
    _Float16* sQ = (_Float16*)(smem + SM_Q);
    _Float16* sK = (_Float16*)(smem + SM_K);
    float*    sS = (float*)(smem + SM_S);
    _Float16* sP = (_Float16*)(smem + SM_P);

    const int tid = threadIdx.x;
    const int lane = tid & 31, wave = tid >> 5;
    const int laneLo = lane & 15, laneHi = lane >> 4;
    const int waveM = wave >> 1, waveN = wave & 1;

    const int id = blockIdx.x;
    const int qb = id & 3;
    const int h  = (id >> 2) & 15;
    const int n  = (id >> 6) & 15;
    const int b  = id >> 10;
    const long headBase = (((long)b * 16 + n) * 16 + h) * (long)(LSEG * HD);
    const _Float16* Q = q16 + headBase;
    const _Float16* K = k16 + headBase;
    const _Float16* V = v16 + headBase;

#if HAVE_TDM
    if (wave == 0) {
        tdm_load_tile(lds_addr_u32(sQ), Q + qb * QB * HD, HD, QB, HD, 5, 3);
        tdm_load_tile(lds_addr_u32(sK), K, HD, LSEG, HD, 5, 3);
        __builtin_amdgcn_s_wait_tensorcnt(0);
    }
#else
    for (int g = tid; g < QB * (HD / 8); g += 256) {
        const int r = g >> 4, c8 = (g & 15) * 8;
        *(v8h*)(sQ + r * LDQ + c8) = *(const v8h*)(Q + (qb * QB + r) * HD + c8);
    }
    for (int g = tid; g < LSEG * (HD / 8); g += 256) {
        const int r = g >> 4, c8 = (g & 15) * 8;
        *(v8h*)(sK + r * LDQ + c8) = *(const v8h*)(K + r * HD + c8);
    }
#endif
    __syncthreads();

    // GEMM1: S = (Q Kt) * scale ; wave tile = 16 rows x 128 keys
    {
        v8f acc[8] = {};
        for (int k0 = 0; k0 < HD; k0 += 32) {
            v16h a = frag_a(sQ + (waveM * 16 + laneLo) * LDQ + k0, laneHi);
#pragma unroll
            for (int j = 0; j < 8; ++j) {
                v16h bf =
                    frag_b(sK + (waveN * 128 + j * 16 + laneLo) * LDQ + k0, laneHi);
                acc[j] = wmma16(a, bf, acc[j]);
            }
        }
        const float scale = 0.088388347648318447f;  // 1/sqrt(128)
#pragma unroll
        for (int j = 0; j < 8; ++j) {
            const int nn = waveN * 128 + j * 16 + laneLo;
            const int m0 = waveM * 16 + laneHi * 8;
#pragma unroll
            for (int i = 0; i < 8; ++i)
                sS[(m0 + i) * LDS_S + nn] = acc[j][i] * scale;
        }
    }
    __syncthreads();

    // stage V transposed into the (now free) sK region: sVt[c][k]
    _Float16* sVt = sK;
    for (int g = tid; g < LSEG * (HD / 4); g += 256) {
        const int r = g >> 5, c4 = (g & 31) * 4;
        v4h f = *(const v4h*)(V + r * HD + c4);
        sVt[(c4 + 0) * LDP + r] = f[0];
        sVt[(c4 + 1) * LDP + r] = f[1];
        sVt[(c4 + 2) * LDP + r] = f[2];
        sVt[(c4 + 3) * LDP + r] = f[3];
    }
    // exact softmax, one row per thread
    if (tid < QB) {
        float mx = -1e30f;
        for (int j = 0; j < LSEG; ++j) mx = fmaxf(mx, sS[tid * LDS_S + j]);
        float sum = 0.f;
        for (int j = 0; j < LSEG; ++j) {
            float e = __expf(sS[tid * LDS_S + j] - mx);
            sum += e;
            sP[tid * LDP + j] = (_Float16)e;
        }
        const float inv = 1.0f / sum;
        for (int j = 0; j < LSEG; ++j)
            sP[tid * LDP + j] = (_Float16)((float)sP[tid * LDP + j] * inv);
    }
    __syncthreads();

    // GEMM2: O = P V ; wave tile = 16 rows x 64 cols, K = 256
    {
        v8f acc[4] = {};
        for (int k0 = 0; k0 < LSEG; k0 += 32) {
            v16h a = frag_a(sP + (waveM * 16 + laneLo) * LDP + k0, laneHi);
#pragma unroll
            for (int j = 0; j < 4; ++j) {
                v16h bf =
                    frag_b(sVt + (waveN * 64 + j * 16 + laneLo) * LDP + k0, laneHi);
                acc[j] = wmma16(a, bf, acc[j]);
            }
        }
        const long rowBase = ((long)b * 16 + n) * 256;
#pragma unroll
        for (int j = 0; j < 4; ++j) {
            const int c  = waveN * 64 + j * 16 + laneLo;
            const int l0 = qb * QB + waveM * 16 + laneHi * 8;
#pragma unroll
            for (int i = 0; i < 8; ++i)
                o16[(rowBase + l0 + i) * (long)DM + h * HD + c] =
                    (_Float16)acc[j][i];
        }
    }
}

// ---------------------------------------------------------------------------
// Kernel 3: out = attn_out @ Wout^T + b_out  (fp32 result)
// ---------------------------------------------------------------------------
__global__ __launch_bounds__(256)
void outproj_gemm(const _Float16* __restrict__ o16, const _Float16* __restrict__ wo16,
                  const float* __restrict__ bout, float* __restrict__ out) {
    __shared__ _Float16 sA[2 * BUFH];
    __shared__ _Float16 sB[2 * BUFH];
    const int tid = threadIdx.x;
    const int lane = tid & 31, wave = tid >> 5;
    const int laneLo = lane & 15, laneHi = lane >> 4;
    const int waveM = wave >> 1, waveN = wave & 1;
    const int mTile = blockIdx.y * 128, nTile = blockIdx.x * 128;

    v8f acc[2][4] = {};
    gemm_tiles_f16(o16 + (long)mTile * DM, wo16 + (long)nTile * DM, DM, DM, DM,
                   sA, sB, acc);

#pragma unroll
    for (int im = 0; im < 2; ++im) {
        const int m0 = mTile + waveM * 32 + im * 16 + laneHi * 8;
#pragma unroll
        for (int jn = 0; jn < 4; ++jn) {
            const int e = nTile + waveN * 64 + jn * 16 + laneLo;
            const float bias = bout[e];
#pragma unroll
            for (int i = 0; i < 8; ++i)
                out[(long)(m0 + i) * DM + e] = acc[im][jn][i] + bias;
        }
    }
}

// ---------------------------------------------------------------------------
extern "C" void kernel_launch(void* const* d_in, const int* in_sizes, int n_in,
                              void* d_out, int out_size, void* d_ws, size_t ws_size,
                              hipStream_t stream) {
    const float* x    = (const float*)d_in[0];
    const float* Wqkv = (const float*)d_in[1];
    const float* bqkv = (const float*)d_in[2];
    const float* Wout = (const float*)d_in[3];
    const float* bout = (const float*)d_in[4];
    float* out = (float*)d_out;

    const size_t MAT = (size_t)NROWS * DM;       // 16.8M elems
    _Float16* xs16 = (_Float16*)d_ws;
    _Float16* wq16 = xs16 + MAT;                 // 6144*2048
    _Float16* wo16 = wq16 + (size_t)QKV_N * DM;  // 2048*2048
    _Float16* q16  = wo16 + (size_t)DM * DM;
    _Float16* k16  = q16 + MAT;
    _Float16* v16  = k16 + MAT;
    _Float16* o16  = v16 + MAT;                  // total ~201 MB

    dim3 blk(256);
    cvt_gather_x<<<dim3((unsigned)(MAT / 4 / 256)), blk, 0, stream>>>(x, xs16);
    cvt_f32_f16<<<dim3((unsigned)((size_t)QKV_N * DM / 4 / 256)), blk, 0, stream>>>(
        Wqkv, wq16);
    cvt_f32_f16<<<dim3((unsigned)((size_t)DM * DM / 4 / 256)), blk, 0, stream>>>(
        Wout, wo16);
    qkv_gemm<<<dim3(QKV_N / 128, NROWS / 128), blk, 0, stream>>>(
        xs16, wq16, bqkv, q16, k16, v16);
    attn_kernel<<<dim3(2 * 16 * 16 * 4), blk, SM_SZ, stream>>>(q16, k16, v16, o16);
    outproj_gemm<<<dim3(DM / 128, NROWS / 128), blk, 0, stream>>>(
        o16, wo16, bout, out);
}
